// PeriodicNeuralWavefunction_80229989089988
// MI455X (gfx1250) — matrix operand order repaired and checked by reading, not compiled, not run on hardware
//
#include <hip/hip_runtime.h>
#include <math.h>

typedef __attribute__((ext_vector_type(16))) _Float16 v16h;
typedef __attribute__((ext_vector_type(8)))  _Float16 v8h;
typedef __attribute__((ext_vector_type(8)))  float    v8f;
typedef __attribute__((ext_vector_type(4)))  unsigned int u32x4;
typedef __attribute__((ext_vector_type(8)))  int      i32x8;
typedef __attribute__((ext_vector_type(4)))  int      i32x4;

#define NE   32
#define DD   64
#define KDET 16

// 16-bit A-matrix 16x32 element->K mapping (contiguous, matches B staging):
// element e (0..15), lane half -> K = e + 16*half.
__device__ __forceinline__ int a_k_idx(int e, int half) {
  return e + (half << 4);
}

// Inverse of the A mapping: for k in [0,32) give (half, e) s.t. a_k_idx(e,half)==k.
__device__ __forceinline__ void a_inv(int k, int& hf, int& e) {
  hf = k >> 4;
  e = k & 15;
}

__device__ __forceinline__ float silu_f(float x) {
  return x / (1.0f + __expf(-x));
}

#define FRAG16(ptr) __builtin_shufflevector((ptr)[0], (ptr)[1], \
    0,1,2,3,4,5,6,7,8,9,10,11,12,13,14,15)

__global__ __launch_bounds__(256)
void wavefn_kernel(const float* __restrict__ r_eg,
                   const float* __restrict__ cellg,
                   const float* __restrict__ Weg,  const float* __restrict__ beg,
                   const float* __restrict__ Wpwg, const float* __restrict__ bpwg,
                   const float* __restrict__ W1g,  const float* __restrict__ b1g,
                   const float* __restrict__ W2g,  const float* __restrict__ b2g,
                   const float* __restrict__ lnwg, const float* __restrict__ lnbg,
                   const float* __restrict__ Wupg, const float* __restrict__ bupg,
                   const float* __restrict__ Wdng, const float* __restrict__ bdng,
                   const float* __restrict__ detwg,
                   const float* __restrict__ Wj1g, const float* __restrict__ bj1g,
                   const float* __restrict__ Wj2g, const float* __restrict__ bj2g,
                   const float* __restrict__ Wj3g, const float* __restrict__ bj3g,
                   const float* __restrict__ jscaleg,
                   float* __restrict__ outg, int n_w)
{
  __shared__ float     s_h[NE][DD];            // current h               (8 KB)
  __shared__ float     s_t[NE][DD];            // h+delta scratch         (8 KB)
  __shared__ _Float16  s_uswz[2 * 2 * NE * 16]; // u in A-fragment order  (4 KB)
  __shared__ __align__(16) unsigned char s_big[32 * 16 * 16 * 4]; // 32 KB W-stage/phi
  __shared__ float     s_hm[DD];
  __shared__ __align__(16) float s_rraw[NE * 3]; // raw positions (TDM target)
  __shared__ float     s_r[NE][3];
  __shared__ float     s_fs[NE][3];
  __shared__ float     s_linv[9];
  __shared__ float     s_cellm[9];
  __shared__ float     s_mu[NE], s_rs[NE];
  __shared__ float     s_ldet[32], s_sig[32];
  __shared__ float     s_wj1[32], s_bj1[32], s_bj2[32], s_wj3[32];
  __shared__ float     s_jw2[32 * 32];
  __shared__ float     s_jsum;

  _Float16* s_w = (_Float16*)s_big;                  // <=12288 f16, B-fragment order
  float (*s_phi)[16][16] = (float (*)[16][16])s_big; // 32 determinant matrices

  const int tid  = threadIdx.x;
  const int lane = tid & 31;
  const int wv   = tid >> 5;       // wave id 0..7
  const int half = lane >> 4;      // 0/1
  const int nl   = lane & 15;      // lane within half
  const int blk  = blockIdx.x;

  // ---------- init: L_inv, cell copy, jastrow staging, TDM position fetch ----------
  if (tid == 0) {
    float c00=cellg[0],c01=cellg[1],c02=cellg[2];
    float c10=cellg[3],c11=cellg[4],c12=cellg[5];
    float c20=cellg[6],c21=cellg[7],c22=cellg[8];
    float det = c00*(c11*c22-c12*c21) - c01*(c10*c22-c12*c20) + c02*(c10*c21-c11*c20);
    float id  = 1.0f / det;
    s_linv[0]=(c11*c22-c12*c21)*id; s_linv[1]=(c02*c21-c01*c22)*id; s_linv[2]=(c01*c12-c02*c11)*id;
    s_linv[3]=(c12*c20-c10*c22)*id; s_linv[4]=(c00*c22-c02*c20)*id; s_linv[5]=(c02*c10-c00*c12)*id;
    s_linv[6]=(c10*c21-c11*c20)*id; s_linv[7]=(c01*c20-c00*c21)*id; s_linv[8]=(c00*c11-c01*c10)*id;
    #pragma unroll
    for (int i = 0; i < 9; ++i) s_cellm[i] = cellg[i];
    s_jsum = 0.0f;
    __builtin_prefetch(W1g, 0, 0);
    __builtin_prefetch(Wupg, 0, 0);
    __builtin_prefetch(Wdng, 0, 0);
  }

#if __has_builtin(__builtin_amdgcn_tensor_load_to_lds) && __has_builtin(__builtin_amdgcn_s_wait_tensorcnt)
  if (tid == 0) {
    // TDM: DMA the walker's 96-float position block into LDS (1-D tile).
    unsigned long long ga = (unsigned long long)(const void*)(r_eg + (size_t)blk * NE * 3);
    unsigned int lds_off = (unsigned int)(size_t)(void*)s_rraw; // flat LDS addr[31:0] == offset
    u32x4 g0;
    g0[0] = 1u;                                   // count=1, user descriptor
    g0[1] = lds_off;                              // lds_addr
    g0[2] = (unsigned int)ga;                     // global_addr[31:0]
    g0[3] = (unsigned int)((ga >> 32) & 0x01FFFFFFu) | 0x80000000u; // addr[56:32] | type=2
    i32x8 g1;
    g1[0] = 0x00020000;                           // workgroup_mask=0, data_size=4B
    g1[1] = (NE * 3) << 16;                       // tensor_dim0 = 96 (bits 63:48)
    g1[2] = 0;                                    // tensor_dim0 hi / tensor_dim1 lo = 0
    g1[3] = (NE * 3) << 16;                       // tile_dim0 = 96 (bits 127:112)
    g1[4] = 0;                                    // tile_dim1 = tile_dim2 = 0 (unused)
    g1[5] = NE * 3;                               // tensor_dim0_stride = 96
    g1[6] = 0;
    g1[7] = 0;
    i32x4 gz4 = {0, 0, 0, 0};
    i32x8 gz8 = {0, 0, 0, 0, 0, 0, 0, 0};
    // 6-arg flavor (clang-23 / therock-10.0 headers): groups 2,3 + extension group.
    __builtin_amdgcn_tensor_load_to_lds(g0, g1, gz4, gz4, gz8, 0);
    __builtin_amdgcn_s_wait_tensorcnt((unsigned short)0);
  }
#else
  if (tid < NE * 3) s_rraw[tid] = r_eg[(size_t)blk * NE * 3 + tid];
#endif

  if (tid < 32) {
    s_wj1[tid] = Wj1g[tid]; s_bj1[tid] = bj1g[tid];
    s_bj2[tid] = bj2g[tid]; s_wj3[tid] = Wj3g[tid];
  }
  for (int i = tid; i < 1024; i += 256) s_jw2[i] = Wj2g[i];
  __syncthreads();

  // ---------- wrap electrons to cell ----------
  if (tid < NE) {
    float x = s_rraw[tid * 3 + 0];
    float y = s_rraw[tid * 3 + 1];
    float z = s_rraw[tid * 3 + 2];
    float f0 = s_linv[0]*x + s_linv[1]*y + s_linv[2]*z;
    float f1 = s_linv[3]*x + s_linv[4]*y + s_linv[5]*z;
    float f2 = s_linv[6]*x + s_linv[7]*y + s_linv[8]*z;
    f0 -= floorf(f0); f1 -= floorf(f1); f2 -= floorf(f2);
    s_fs[tid][0] = f0; s_fs[tid][1] = f1; s_fs[tid][2] = f2;
    s_r[tid][0] = f0*s_cellm[0] + f1*s_cellm[3] + f2*s_cellm[6];
    s_r[tid][1] = f0*s_cellm[1] + f1*s_cellm[4] + f2*s_cellm[7];
    s_r[tid][2] = f0*s_cellm[2] + f1*s_cellm[5] + f2*s_cellm[8];
  }
  __syncthreads();

  // ---------- embed: h = r@W_e + b_e + pw@W_pw + b_pw ----------
  const float TWO_PI = 6.28318530717958647692f;
  for (int idx = tid; idx < NE * DD; idx += 256) {
    int e = idx >> 6, d = idx & 63;
    float acc = beg[d] + bpwg[d];
    acc += s_r[e][0]*Weg[0*DD+d] + s_r[e][1]*Weg[1*DD+d] + s_r[e][2]*Weg[2*DD+d];
    #pragma unroll
    for (int c = 0; c < 3; ++c) {
      float sn, cs;
      __sincosf(TWO_PI * s_fs[e][c], &sn, &cs);
      acc += sn * Wpwg[c*DD + d] + cs * Wpwg[(c+3)*DD + d];
    }
    s_h[e][d] = acc;
  }

  // ---------- interaction layers ----------
  const int mi = wv & 1;          // M tile (0/1)
  const int ni = wv >> 1;         // N tile (0..3)
  for (int l = 0; l < 3; ++l) {
    __syncthreads();
    if (tid < DD) {               // hm = mean over electrons
      float a = 0.f;
      #pragma unroll
      for (int e = 0; e < NE; ++e) a += s_h[e][tid];
      s_hm[tid] = a * (1.0f / NE);
    }
    // stage W1[l] as f16 directly in B-fragment order:
    // dest[((k>>4)*64 + n)*16 + (k&15)]  (B element e == k mod 16 per 16-k block)
    for (int idx = tid; idx < 192 * 64; idx += 256) {
      int k = idx >> 6, n = idx & 63;
      s_w[(((k >> 4) * 64) + n) * 16 + (k & 15)] = (_Float16)W1g[l * 192 * 64 + idx];
    }
    __syncthreads();

    // u = silu( [h, hm, h*hm] @ W1 + b1 )   M=32 K=192 N=64
    {
      const int m = mi * 16 + nl;
      v8f acc = {};
      for (int kc = 0; kc < 6; ++kc) {
        v16h A;
        #pragma unroll
        for (int e = 0; e < 16; ++e) {
          int k = kc * 32 + a_k_idx(e, half);
          float xv;
          if (k < 64)       xv = s_h[m][k];
          else if (k < 128) xv = s_hm[k - 64];
          else              xv = s_h[m][k - 128] * s_hm[k - 128];
          A[e] = (_Float16)xv;
        }
        const v8h* bp = (const v8h*)&s_w[((kc * 2 + half) * 64 + ni * 16 + nl) * 16];
        v16h B = FRAG16(bp);
        acc = __builtin_amdgcn_wmma_f32_16x16x32_f16(false, A, false, B,
                                                     (short)0, acc, false, false);
      }
      // epilogue: bias+silu, write u in A-fragment order for matmul-2
      int hf2, e2;
      int nn = ni * 16 + nl;
      a_inv(nn & 31, hf2, e2);
      int kc2 = nn >> 5;
      #pragma unroll
      for (int j = 0; j < 8; ++j) {
        int mm = mi * 16 + half * 8 + j;
        float uv = silu_f(acc[j] + b1g[l * 64 + nn]);
        s_uswz[((kc2 * 2 + hf2) * NE + mm) * 16 + e2] = (_Float16)uv;
      }
    }
    __syncthreads();
    for (int idx = tid; idx < 64 * 64; idx += 256) {   // stage W2[l], B-fragment order
      int k = idx >> 6, n = idx & 63;
      s_w[(((k >> 4) * 64) + n) * 16 + (k & 15)] = (_Float16)W2g[l * 64 * 64 + idx];
    }
    __syncthreads();

    // t = h + u @ W2 + b2    M=32 K=64 N=64  (both operands vector-loaded)
    {
      const int m = mi * 16 + nl;
      v8f acc = {};
      #pragma unroll
      for (int kc = 0; kc < 2; ++kc) {
        const v8h* ap = (const v8h*)&s_uswz[((kc * 2 + half) * NE + m) * 16];
        v16h A = FRAG16(ap);
        const v8h* bp = (const v8h*)&s_w[((kc * 2 + half) * 64 + ni * 16 + nl) * 16];
        v16h B = FRAG16(bp);
        acc = __builtin_amdgcn_wmma_f32_16x16x32_f16(false, A, false, B,
                                                     (short)0, acc, false, false);
      }
      #pragma unroll
      for (int j = 0; j < 8; ++j) {
        int mm = mi * 16 + half * 8 + j;
        int nn = ni * 16 + nl;
        s_t[mm][nn] = s_h[mm][nn] + acc[j] + b2g[l * 64 + nn];
      }
    }
    __syncthreads();

    // layernorm rows of s_t -> s_h
    if (tid < NE) {
      float mu = 0.f;
      #pragma unroll
      for (int d = 0; d < DD; ++d) mu += s_t[tid][d];
      mu *= (1.0f / DD);
      float v = 0.f;
      #pragma unroll
      for (int d = 0; d < DD; ++d) { float df = s_t[tid][d] - mu; v += df * df; }
      s_mu[tid] = mu;
      s_rs[tid] = rsqrtf(v * (1.0f / DD) + 1e-5f);
    }
    __syncthreads();
    for (int idx = tid; idx < NE * DD; idx += 256) {
      int e = idx >> 6, d = idx & 63;
      s_h[e][d] = (s_t[e][d] - s_mu[e]) * s_rs[e] * lnwg[l * 64 + d] + lnbg[l * 64 + d];
    }
  }
  __syncthreads();

  // ---------- orbitals: phi[t] = h[rows] @ W[:, 16t:16t+16] + b  (32 tiles) ----------
  for (int it = 0; it < 4; ++it) {
    int t = wv * 4 + it;                     // 0..15 up dets, 16..31 dn dets
    bool up = (t < 16);
    int kdet = up ? t : (t - 16);
    const float* W  = up ? Wupg : Wdng;
    const float* bb = up ? bupg : bdng;
    int rowoff = up ? 0 : 16;
    v8f acc = {};
    #pragma unroll
    for (int kc = 0; kc < 2; ++kc) {
      v16h A, B;
      #pragma unroll
      for (int e = 0; e < 16; ++e) {
        int k = kc * 32 + a_k_idx(e, half);
        A[e] = (_Float16)s_h[rowoff + nl][k];
        int kb = kc * 32 + e + half * 16;
        B[e] = (_Float16)W[kb * 256 + kdet * 16 + nl];
      }
      acc = __builtin_amdgcn_wmma_f32_16x16x32_f16(false, A, false, B,
                                                   (short)0, acc, false, false);
    }
    #pragma unroll
    for (int j = 0; j < 8; ++j) {
      int mm = half * 8 + j;
      s_phi[t][mm][nl] = acc[j] + bb[kdet * 16 + nl];
    }
  }
  __syncthreads();

  // ---------- slogdet: one 16x16 LU (partial pivoting) per lane of wave 0 ----------
  if (tid < 32) {
    float (*A)[16] = s_phi[tid];
    float sgn = 1.0f, ld = 0.0f;
    for (int c = 0; c < 16; ++c) {
      int piv = c; float pv = fabsf(A[c][c]);
      for (int rr = c + 1; rr < 16; ++rr) {
        float av = fabsf(A[rr][c]);
        if (av > pv) { pv = av; piv = rr; }
      }
      if (piv != c) {
        sgn = -sgn;
        for (int cc = 0; cc < 16; ++cc) {
          float tmp = A[c][cc]; A[c][cc] = A[piv][cc]; A[piv][cc] = tmp;
        }
      }
      float d = A[c][c];
      if (d < 0.0f) sgn = -sgn;
      ld += __logf(fabsf(d) + 1e-30f);
      float inv = 1.0f / d;
      for (int rr = c + 1; rr < 16; ++rr) {
        float f = A[rr][c] * inv;
        for (int cc = c + 1; cc < 16; ++cc) A[rr][cc] -= f * A[c][cc];
      }
    }
    s_ldet[tid] = ld;
    s_sig[tid]  = sgn;
  }

  // ---------- Jastrow: 496 pairs, MLP 1->32->32->1 ----------
  {
    float jloc = 0.0f;
    float bj3v = bj3g[0];
    for (int p = tid; p < 496; p += 256) {
      int i = 0, rem = p;
      while (rem >= 31 - i) { rem -= 31 - i; ++i; }
      int jx = i + 1 + rem;
      float dx = s_r[i][0] - s_r[jx][0];
      float dy = s_r[i][1] - s_r[jx][1];
      float dz = s_r[i][2] - s_r[jx][2];
      float q0 = s_linv[0]*dx + s_linv[1]*dy + s_linv[2]*dz;
      float q1 = s_linv[3]*dx + s_linv[4]*dy + s_linv[5]*dz;
      float q2 = s_linv[6]*dx + s_linv[7]*dy + s_linv[8]*dz;
      q0 -= rintf(q0); q1 -= rintf(q1); q2 -= rintf(q2);
      float vx = q0*s_cellm[0] + q1*s_cellm[3] + q2*s_cellm[6];
      float vy = q0*s_cellm[1] + q1*s_cellm[4] + q2*s_cellm[7];
      float vz = q0*s_cellm[2] + q1*s_cellm[5] + q2*s_cellm[8];
      float rij = sqrtf(vx*vx + vy*vy + vz*vz);
      float a1[32];
      #pragma unroll
      for (int k = 0; k < 32; ++k) a1[k] = silu_f(rij * s_wj1[k] + s_bj1[k]);
      float o = bj3v;
      for (int k = 0; k < 32; ++k) {
        float v = s_bj2[k];
        #pragma unroll
        for (int m2 = 0; m2 < 32; ++m2) v += a1[m2] * s_jw2[m2 * 32 + k];
        o += silu_f(v) * s_wj3[k];
      }
      jloc += o;
    }
    atomicAdd(&s_jsum, jloc);
  }
  __syncthreads();

  // ---------- signed logsumexp combine ----------
  if (tid == 0) {
    float mx = -1e30f;
    for (int k = 0; k < KDET; ++k) mx = fmaxf(mx, detwg[k]);
    float se = 0.0f;
    for (int k = 0; k < KDET; ++k) se += __expf(detwg[k] - mx);
    float logs[KDET], sg[KDET];
    float m2 = -1e30f;
    for (int k = 0; k < KDET; ++k) {
      float sm = __expf(detwg[k] - mx) / se;
      logs[k] = s_ldet[k] + s_ldet[16 + k] + __logf(sm + 1e-10f);
      sg[k] = s_sig[k] * s_sig[16 + k];
      m2 = fmaxf(m2, logs[k]);
    }
    float ssum = 0.0f;
    for (int k = 0; k < KDET; ++k) ssum += sg[k] * __expf(logs[k] - m2);
    float lp = jscaleg[0] * s_jsum + m2 + __logf(fabsf(ssum) + 1e-30f);
    float sp = (ssum > 0.0f) ? 1.0f : ((ssum < 0.0f) ? -1.0f : 0.0f);
    outg[blk]        = lp;
    outg[n_w + blk]  = sp;
  }
}

extern "C" void kernel_launch(void* const* d_in, const int* in_sizes, int n_in,
                              void* d_out, int out_size, void* d_ws, size_t ws_size,
                              hipStream_t stream) {
  const float* r_e   = (const float*)d_in[0];
  const float* cell  = (const float*)d_in[1];
  // d_in[2] = twist (unused by reference math)
  const float* W_e   = (const float*)d_in[3];
  const float* b_e   = (const float*)d_in[4];
  const float* W_pw  = (const float*)d_in[5];
  const float* b_pw  = (const float*)d_in[6];
  const float* W1    = (const float*)d_in[7];
  const float* b1    = (const float*)d_in[8];
  const float* W2    = (const float*)d_in[9];
  const float* b2    = (const float*)d_in[10];
  const float* ln_w  = (const float*)d_in[11];
  const float* ln_b  = (const float*)d_in[12];
  const float* W_up  = (const float*)d_in[13];
  const float* b_up  = (const float*)d_in[14];
  const float* W_dn  = (const float*)d_in[15];
  const float* b_dn  = (const float*)d_in[16];
  const float* detw  = (const float*)d_in[17];
  const float* Wj1   = (const float*)d_in[18];
  const float* bj1   = (const float*)d_in[19];
  const float* Wj2   = (const float*)d_in[20];
  const float* bj2   = (const float*)d_in[21];
  const float* Wj3   = (const float*)d_in[22];
  const float* bj3   = (const float*)d_in[23];
  const float* jsc   = (const float*)d_in[24];
  float* out = (float*)d_out;

  int n_w = in_sizes[0] / (NE * 3);   // 4096 walkers
  hipLaunchKernelGGL(wavefn_kernel, dim3(n_w), dim3(256), 0, stream,
                     r_e, cell, W_e, b_e, W_pw, b_pw, W1, b1, W2, b2,
                     ln_w, ln_b, W_up, b_up, W_dn, b_dn, detw,
                     Wj1, bj1, Wj2, bj2, Wj3, bj3, jsc, out, n_w);
}
